// GIN_Layer_72688026518105
// MI455X (gfx1250) — matrix operand order repaired and checked by reading, not compiled, not run on hardware
//
#include <hip/hip_runtime.h>
#include <hip/hip_bf16.h>

typedef __attribute__((ext_vector_type(16))) _Float16 v16h;
typedef __attribute__((ext_vector_type(8)))  _Float16 v8h;
typedef __attribute__((ext_vector_type(8)))  float    v8f;

// ---------------------------------------------------------------------------
// K0: repack a 128x128 fp32 weight (row-major, W[k][n]) into f16 WMMA
// B-fragment layout. B 32x16 f16 layout (wave32): lane l holds column
// n = 16t + (l&15), kb = 32s + (l>>4)*8, VGPR i = pack(K=kb+2i+(i>=4?8:0), +1).
// Stored so each lane's 8 dwords are contiguous (32B): dword index
// ((t*4+s)*32 + l)*8 + i.
// ---------------------------------------------------------------------------
__global__ void pack_weights(const float* __restrict__ W, _Float16* __restrict__ dst) {
  int tid = blockIdx.x * 256 + threadIdx.x;          // 8192 dwords total
  if (tid >= 8192) return;
  int i = tid & 7;
  int l = (tid >> 3) & 31;
  int s = (tid >> 8) & 3;
  int t = (tid >> 10) & 7;
  int n = 16 * t + (l & 15);
  int k = 32 * s + (l >> 4) * 8 + 2 * i + (i >= 4 ? 8 : 0);
  dst[2 * tid]     = (_Float16)W[k * 128 + n];
  dst[2 * tid + 1] = (_Float16)W[(k + 1) * 128 + n];
}

// ---------------------------------------------------------------------------
// K1: h0 = (1+eps)*node ; also zero the 192 per-graph stat floats.
// ---------------------------------------------------------------------------
__global__ void init_h0(const float4* __restrict__ node4, float4* __restrict__ h0_4,
                        const float* __restrict__ eps_p, float* __restrict__ stats,
                        long long n4) {
  long long i = (long long)blockIdx.x * 256 + threadIdx.x;
  if (i < 192) stats[i] = 0.0f;
  if (i >= n4) return;
  float s = 1.0f + eps_p[0];
  float4 v = node4[i];
  v.x *= s; v.y *= s; v.z *= s; v.w *= s;
  h0_4[i] = v;
}

// ---------------------------------------------------------------------------
// K2: GIN sum aggregation. One wave per edge: 32 lanes x float4 = 128 feats.
// h0[dst] += node[src]  via hardware f32 atomics.
// ---------------------------------------------------------------------------
__global__ void edge_scatter(const float4* __restrict__ node4, const int* __restrict__ ei,
                             float* __restrict__ h0, int E) {
  int e = blockIdx.x * 8 + (threadIdx.x >> 5);
  if (e >= E) return;
  int lane = threadIdx.x & 31;
  int src = ei[e];
  int dst = ei[E + e];
  float4 v = node4[(long long)src * 32 + lane];
  float* p = h0 + (long long)dst * 128 + lane * 4;
  unsafeAtomicAdd(p + 0, v.x);
  unsafeAtomicAdd(p + 1, v.y);
  unsafeAtomicAdd(p + 2, v.z);
  unsafeAtomicAdd(p + 3, v.w);
}

// ---------------------------------------------------------------------------
// Fused-MLP helpers (per wave: 16 rows x 128 cols = 8 C tiles).
// ---------------------------------------------------------------------------
__device__ __forceinline__ void gemm128(const v16h a[4], const _Float16* __restrict__ wp,
                                        int lane, v8f acc[8]) {
#pragma unroll
  for (int t = 0; t < 8; ++t) {
    v8f c = {};
#pragma unroll
    for (int s = 0; s < 4; ++s) {
      v16h b = *(const v16h*)(wp + ((size_t)((t * 4 + s) * 32 + lane)) * 16);
      c = __builtin_amdgcn_wmma_f32_16x16x32_f16(false, a[s], false, b,
                                                 (short)0, c, false, false);
    }
    acc[t] = c;
  }
}

// bias + row LayerNorm + ReLU, write f16 row-major tile into per-wave LDS.
__device__ __forceinline__ void ln_relu_to_lds(v8f acc[8], const float* __restrict__ bias,
                                               const float* __restrict__ g,
                                               const float* __restrict__ be,
                                               _Float16* __restrict__ lbase, int lane) {
  int m = lane & 15, half = lane >> 4;
#pragma unroll
  for (int t = 0; t < 8; ++t) {
    float bv = bias[16 * t + m];
#pragma unroll
    for (int j = 0; j < 8; ++j) acc[t][j] += bv;
  }
  float mu[8], rstd[8];
#pragma unroll
  for (int j = 0; j < 8; ++j) {
    float s = 0.f, q = 0.f;
#pragma unroll
    for (int t = 0; t < 8; ++t) { float v = acc[t][j]; s += v; q += v * v; }
#pragma unroll
    for (int msk = 1; msk < 16; msk <<= 1) {   // stays within 16-lane half
      s += __shfl_xor(s, msk, 32);
      q += __shfl_xor(q, msk, 32);
    }
    float mmu = s * (1.0f / 128.0f);
    float var = q * (1.0f / 128.0f) - mmu * mmu;
    var = var > 0.f ? var : 0.f;
    mu[j] = mmu;
    rstd[j] = rsqrtf(var + 1e-5f);
  }
#pragma unroll
  for (int t = 0; t < 8; ++t) {
    float gv = g[16 * t + m], bev = be[16 * t + m];
#pragma unroll
    for (int j = 0; j < 8; ++j) {
      float y = (acc[t][j] - mu[j]) * rstd[j] * gv + bev;
      y = y > 0.f ? y : 0.f;
      lbase[(half * 8 + j) * 128 + 16 * t + m] = (_Float16)y;
    }
  }
  asm volatile("" ::: "memory");   // keep DS loads after DS stores
}

// Load f16 A-fragments from per-wave LDS row tile (row-major 16x128).
__device__ __forceinline__ void load_a_lds(const _Float16* __restrict__ lbase, int lane,
                                           v16h a[4]) {
  int m = lane & 15, half = lane >> 4;
  const _Float16* rp = lbase + m * 128;
#pragma unroll
  for (int s = 0; s < 4; ++s) {
    int kb = 32 * s + half * 8;
    v8h lo = *(const v8h*)(rp + kb);
    v8h hi = *(const v8h*)(rp + kb + 16);
    a[s] = __builtin_shufflevector(lo, hi, 0, 1, 2, 3, 4, 5, 6, 7,
                                   8, 9, 10, 11, 12, 13, 14, 15);
  }
}

// ---------------------------------------------------------------------------
// K3: fused 3-layer MLP. 8 waves/block, wave handles rows [rowbase, rowbase+16).
// 96 v_wmma per wave. Final layer -> f32 to d_out + per-graph sum/sumsq atomics.
// ---------------------------------------------------------------------------
__global__ __launch_bounds__(256, 1) void gin_mlp(
    const float* __restrict__ h0,
    const _Float16* __restrict__ wp1, const _Float16* __restrict__ wp2,
    const _Float16* __restrict__ wp3,
    const float* __restrict__ b1, const float* __restrict__ g1, const float* __restrict__ be1,
    const float* __restrict__ b2, const float* __restrict__ g2, const float* __restrict__ be2,
    const float* __restrict__ b3,
    const int* __restrict__ bp,
    float* __restrict__ out,
    float* __restrict__ gsum, float* __restrict__ gsumsq, float* __restrict__ gcnt,
    int N) {
  __shared__ __align__(32) _Float16 smem[8 * 16 * 128];   // 32 KB, per-wave 4 KB
  int wave = threadIdx.x >> 5;
  int lane = threadIdx.x & 31;
  int m = lane & 15, half = lane >> 4;
  int rowbase = blockIdx.x * 128 + wave * 16;
  _Float16* lbase = smem + wave * (16 * 128);

  // A fragments for layer 1 straight from global h0 (fp32 -> f16).
  int r = rowbase + m;
  int rc = r < N ? r : N - 1;
  const float* rowp = h0 + (long long)rc * 128;
  v16h a[4];
#pragma unroll
  for (int s = 0; s < 4; ++s) {
    int kb = 32 * s + half * 8;
    float4 x0 = *(const float4*)(rowp + kb);
    float4 x1 = *(const float4*)(rowp + kb + 4);
    float4 x2 = *(const float4*)(rowp + kb + 16);
    float4 x3 = *(const float4*)(rowp + kb + 20);
    v16h av;
    av[0]  = (_Float16)x0.x; av[1]  = (_Float16)x0.y;
    av[2]  = (_Float16)x0.z; av[3]  = (_Float16)x0.w;
    av[4]  = (_Float16)x1.x; av[5]  = (_Float16)x1.y;
    av[6]  = (_Float16)x1.z; av[7]  = (_Float16)x1.w;
    av[8]  = (_Float16)x2.x; av[9]  = (_Float16)x2.y;
    av[10] = (_Float16)x2.z; av[11] = (_Float16)x2.w;
    av[12] = (_Float16)x3.x; av[13] = (_Float16)x3.y;
    av[14] = (_Float16)x3.z; av[15] = (_Float16)x3.w;
    a[s] = av;
  }

  v8f acc[8];
  gemm128(a, wp1, lane, acc);
  ln_relu_to_lds(acc, b1, g1, be1, lbase, lane);

  load_a_lds(lbase, lane, a);
  gemm128(a, wp2, lane, acc);
  ln_relu_to_lds(acc, b2, g2, be2, lbase, lane);

  load_a_lds(lbase, lane, a);
  gemm128(a, wp3, lane, acc);

  // Final bias, row sums for graph-LN stats, masked stores.
#pragma unroll
  for (int t = 0; t < 8; ++t) {
    float bv = b3[16 * t + m];
#pragma unroll
    for (int j = 0; j < 8; ++j) acc[t][j] += bv;
  }
#pragma unroll
  for (int j = 0; j < 8; ++j) {
    float s = 0.f, q = 0.f;
#pragma unroll
    for (int t = 0; t < 8; ++t) { float v = acc[t][j]; s += v; q += v * v; }
#pragma unroll
    for (int msk = 1; msk < 16; msk <<= 1) {
      s += __shfl_xor(s, msk, 32);
      q += __shfl_xor(q, msk, 32);
    }
    int rr = rowbase + half * 8 + j;
    if (rr < N) {
#pragma unroll
      for (int t = 0; t < 8; ++t)
        out[(long long)rr * 128 + 16 * t + m] = acc[t][j];
      if (m == 0) {
        int g = bp[rr];
        unsafeAtomicAdd(gsum + g, s);
        unsafeAtomicAdd(gsumsq + g, q);
        unsafeAtomicAdd(gcnt + g, 1.0f);
      }
    }
  }
}

// ---------------------------------------------------------------------------
// K4: per-graph LayerNorm (graph mode) + affine + ReLU, in-place on d_out.
// var = E[x^2] - mean^2 over all nodes*feats of the graph.
// ---------------------------------------------------------------------------
__global__ void graph_norm(float* __restrict__ out, const int* __restrict__ bp,
                           const float* __restrict__ gsum, const float* __restrict__ gsumsq,
                           const float* __restrict__ gcnt,
                           const float* __restrict__ lnw, const float* __restrict__ lnb,
                           long long n4) {
  long long i = (long long)blockIdx.x * 256 + threadIdx.x;
  if (i >= n4) return;
  int rr = (int)(i >> 5);
  int c4 = ((int)i & 31) * 4;
  int g = bp[rr];
  float norm = gcnt[g] * 128.0f;
  norm = norm > 1.0f ? norm : 1.0f;
  float mean = gsum[g] / norm;
  float var = gsumsq[g] / norm - mean * mean;
  var = var > 0.f ? var : 0.f;
  float rstd = rsqrtf(var + 1e-5f);
  float* p = out + (long long)rr * 128 + c4;
  float4 x = *(float4*)p;
  float4 w = *(const float4*)(lnw + c4);
  float4 b = *(const float4*)(lnb + c4);
  x.x = fmaxf((x.x - mean) * rstd * w.x + b.x, 0.f);
  x.y = fmaxf((x.y - mean) * rstd * w.y + b.y, 0.f);
  x.z = fmaxf((x.z - mean) * rstd * w.z + b.z, 0.f);
  x.w = fmaxf((x.w - mean) * rstd * w.w + b.w, 0.f);
  *(float4*)p = x;
}

// ---------------------------------------------------------------------------
extern "C" void kernel_launch(void* const* d_in, const int* in_sizes, int n_in,
                              void* d_out, int out_size, void* d_ws, size_t ws_size,
                              hipStream_t stream) {
  const float* node = (const float*)d_in[0];
  const int*   ei   = (const int*)d_in[1];
  // d_in[2] = edge_attr (unused by the reference)
  const int*   bp   = (const int*)d_in[3];
  const float* eps  = (const float*)d_in[4];
  const float* W1 = (const float*)d_in[5];  const float* b1 = (const float*)d_in[6];
  const float* g1 = (const float*)d_in[7];  const float* be1 = (const float*)d_in[8];
  const float* W2 = (const float*)d_in[9];  const float* b2 = (const float*)d_in[10];
  const float* g2 = (const float*)d_in[11]; const float* be2 = (const float*)d_in[12];
  const float* W3 = (const float*)d_in[13]; const float* b3 = (const float*)d_in[14];
  const float* lnw = (const float*)d_in[15]; const float* lnb = (const float*)d_in[16];

  int N = in_sizes[0] / 128;
  int E = in_sizes[1] / 2;
  float* out = (float*)d_out;

  // Workspace layout: h0 (N*128 f32) | 3x packed f16 weights (32 KB each) | stats
  char* ws = (char*)d_ws;
  float* h0 = (float*)ws;
  size_t offW = (size_t)N * 128 * sizeof(float);
  _Float16* wp1 = (_Float16*)(ws + offW);
  _Float16* wp2 = wp1 + 16384;
  _Float16* wp3 = wp2 + 16384;
  float* stats = (float*)(ws + offW + 3 * 16384 * sizeof(_Float16));
  float* gsum = stats, *gsumsq = stats + 64, *gcnt = stats + 128;

  pack_weights<<<32, 256, 0, stream>>>(W1, wp1);
  pack_weights<<<32, 256, 0, stream>>>(W2, wp2);
  pack_weights<<<32, 256, 0, stream>>>(W3, wp3);

  long long n4 = (long long)N * 32;
  int blocks1 = (int)((n4 + 255) / 256);
  init_h0<<<blocks1, 256, 0, stream>>>((const float4*)node, (float4*)h0, eps, stats, n4);

  int blocks2 = (E + 7) / 8;
  edge_scatter<<<blocks2, 256, 0, stream>>>((const float4*)node, ei, h0, E);

  int blocks3 = (N + 127) / 128;
  gin_mlp<<<blocks3, 256, 0, stream>>>(h0, wp1, wp2, wp3,
                                       b1, g1, be1, b2, g2, be2, b3,
                                       bp, out, gsum, gsumsq, gcnt, N);

  graph_norm<<<blocks1, 256, 0, stream>>>(out, bp, gsum, gsumsq, gcnt, lnw, lnb, n4);
}
